// MultiwayMatching_25744033973064
// MI455X (gfx1250) — matrix operand order repaired and checked by reading, not compiled, not run on hardware
//
#include <hip/hip_runtime.h>
#include <math.h>

#define DEV __device__ __forceinline__

typedef __bf16 bf16_t;
typedef __bf16 v16bf __attribute__((ext_vector_type(16)));
typedef __bf16 v8bf  __attribute__((ext_vector_type(8)));
typedef __bf16 v4bf  __attribute__((ext_vector_type(4)));
typedef float  v8f   __attribute__((ext_vector_type(8)));
typedef float  v4f   __attribute__((ext_vector_type(4)));

union FragU { v16bf v; v8bf h[2]; };

// Hardware tanh (gfx1250 V_TANH_F32, TRANS32 unit). One independent op
// (v_nop) after a TRANS op before its result is consumed (ISA 7.4).
DEV float tanh_hw(float x) {
  float y;
  asm("v_tanh_f32 %0, %1\n\tv_nop" : "=v"(y) : "v"(x));
  return y;
}

// ---------------------------------------------------------------------------
// WMMA: D(16x16,f32) = A(16x32,bf16) * B(32x16,bf16) + C
//   A: lane r=lane&15 -> row r; half hh=lane>>4 selects K 0..15 / 16..31
//   B: lane n=lane&15 -> col n; half selects K 0..15 / 16..31 (contiguous)
//   C/D: col = lane&15, rows = hh*8 + v
// All B operands are stored TRANSPOSED (Bt[n][k]) so both A and B fragments
// are two contiguous 16-byte loads.
// ---------------------------------------------------------------------------
DEV v8f wmma_bf16(v16bf a, v16bf b, v8f c) {
  return __builtin_amdgcn_wmma_f32_16x16x32_bf16(false, a, false, b,
                                                 (short)0, c, false, false);
}

// ---------------------------------------------------------------------------
// Batched GEMM, one wave per 16x16 C tile, K=256 fully unrolled.
// C[b] (MxN, OutT) = A[b] (MxK, bf16 row-major) @ Bt[b] (NxK, bf16 row-major)^T
// ---------------------------------------------------------------------------
template <typename OutT>
__global__ __launch_bounds__(256) void gemm_bt_k(
    const bf16_t* __restrict__ A, const bf16_t* __restrict__ Bt,
    OutT* __restrict__ C, int tilesM, int tilesN, int nbatch,
    int lda, int ldb, int ldc, long sA, long sB, long sC)
{
  int lane = threadIdx.x & 31;
  int gw   = blockIdx.x * 8 + (threadIdx.x >> 5);
  int tilesPer = tilesM * tilesN;
  if (gw >= tilesPer * nbatch) return;          // wave-uniform exit
  int batch = gw / tilesPer;
  int t     = gw - batch * tilesPer;
  int mt = t / tilesN, nt = t - mt * tilesN;
  int r = lane & 15, hh = lane >> 4;
  const bf16_t* pa = A + (size_t)batch * sA + (size_t)(mt * 16 + r) * lda + hh * 16;
  const bf16_t* pb = Bt + (size_t)batch * sB + (size_t)(nt * 16 + r) * ldb + hh * 16;
  v8f acc = {0.f,0.f,0.f,0.f,0.f,0.f,0.f,0.f};
#pragma unroll
  for (int kb = 0; kb < 256; kb += 32) {
    FragU a, b;
    a.h[0] = *(const v8bf*)(pa + kb);
    a.h[1] = *(const v8bf*)(pa + kb + 8);
    b.h[0] = *(const v8bf*)(pb + kb);
    b.h[1] = *(const v8bf*)(pb + kb + 8);
    acc = wmma_bf16(a.v, b.v, acc);
  }
  OutT* pc = C + (size_t)batch * sC + (size_t)(mt * 16 + hh * 8) * ldc + nt * 16 + r;
#pragma unroll
  for (int v = 0; v < 8; ++v) pc[(size_t)v * ldc] = (OutT)acc[v];
}

// ---------------------------------------------------------------------------
// Vectorized f32 -> bf16 convert (b128 load, b64 store), n multiple of 1024.
__global__ void cvt_bf16_k(const float* __restrict__ x, bf16_t* __restrict__ y,
                           int n4) {
  int i = blockIdx.x * blockDim.x + threadIdx.x;
  if (i >= n4) return;
  v4f a = *(const v4f*)(x + (size_t)i * 4);
  v4bf g;
  g.x = (bf16_t)a.x; g.y = (bf16_t)a.y; g.z = (bf16_t)a.z; g.w = (bf16_t)a.w;
  *(v4bf*)(y + (size_t)i * 4) = g;
}

// Tiled transpose + f32->bf16: y[b][c][r] = x[b][r][c]
__global__ __launch_bounds__(256) void cvt_t_bf16_k(
    const float* __restrict__ x, bf16_t* __restrict__ y,
    int rows, int cols, long sIn, long sOut)
{
  __shared__ float tile[32][33];
  int ctiles = cols >> 5, rtiles = rows >> 5;
  int per = ctiles * rtiles;
  int batch = blockIdx.x / per;
  int t = blockIdx.x - batch * per;
  int rt = t / ctiles, ct = t - rt * ctiles;
  const float* xb = x + (size_t)batch * sIn;
  bf16_t* yb = y + (size_t)batch * sOut;
  int lx = threadIdx.x & 31, ly = threadIdx.x >> 5;   // 32 x 8
#pragma unroll
  for (int i = 0; i < 32; i += 8)
    tile[ly + i][lx] = xb[(size_t)(rt * 32 + ly + i) * cols + ct * 32 + lx];
  __syncthreads();
#pragma unroll
  for (int i = 0; i < 32; i += 8)
    yb[(size_t)(ct * 32 + ly + i) * rows + rt * 32 + lx] = (bf16_t)tile[lx][ly + i];
}

// ---------------------------------------------------------------------------
// sc[b,p,q] = sum_h tanh(S1[b,q,h] + sign*S2[b,p,h]) * v[h]
// 16p x 16q tile per block; rows staged in LDS; float4 LDS reads; hw tanh.
// ---------------------------------------------------------------------------
__global__ __launch_bounds__(256) void addsub_scores_k(
    const float* __restrict__ S1, const float* __restrict__ S2,
    const float* __restrict__ vh, float sign, float* __restrict__ SC)
{
  const int L = 256, Hh = 256;
  __shared__ __align__(16) float s1[16][260];
  __shared__ __align__(16) float s2[16][260];
  __shared__ __align__(16) float vv[256];
  int bidx = blockIdx.x;
  int qt = bidx & 15, pt = (bidx >> 4) & 15, b = bidx >> 8;
  int tid = threadIdx.x;
  for (int idx = tid; idx < 16 * 64; idx += 256) {      // float4 granularity
    int r = idx >> 6, c4 = (idx & 63) << 2;
    *(v4f*)&s1[r][c4] = *(const v4f*)&S1[((size_t)b * L + qt * 16 + r) * Hh + c4];
    v4f u = *(const v4f*)&S2[((size_t)b * L + pt * 16 + r) * Hh + c4];
    u.x *= sign; u.y *= sign; u.z *= sign; u.w *= sign;
    *(v4f*)&s2[r][c4] = u;
  }
  vv[tid] = vh[tid];
  __syncthreads();
  int lq = tid & 15, lp = tid >> 4;
  const v4f* r1 = (const v4f*)&s1[lq][0];
  const v4f* r2 = (const v4f*)&s2[lp][0];
  const v4f* rv = (const v4f*)&vv[0];
  float acc = 0.f;
#pragma unroll 8
  for (int h4 = 0; h4 < Hh / 4; ++h4) {
    v4f a = r1[h4], c = r2[h4], w4 = rv[h4];
    acc += tanh_hw(a.x + c.x) * w4.x;
    acc += tanh_hw(a.y + c.y) * w4.y;
    acc += tanh_hw(a.z + c.z) * w4.z;
    acc += tanh_hw(a.w + c.w) * w4.w;
  }
  SC[((size_t)b * L + pt * 16 + lp) * L + qt * 16 + lq] = acc;
}

// ---------------------------------------------------------------------------
// Branch d: per (b,p) block, S = (Hq .* Hp[p]) @ Wd  (256x256x256 bf16 WMMA),
// fused epilogue sd[q] = sum_h tanh(S[q,h]) * vd[h].
// 8 waves; q in chunks of 64; A frags from LDS G (ds_load_b128), B frags
// straight from global WdT (L2/WGP$-resident). kt loop is barrier-free.
// ---------------------------------------------------------------------------
__global__ __launch_bounds__(256) void branch_d_scores_k(
    const float* __restrict__ Hq, const float* __restrict__ Hp,
    const bf16_t* __restrict__ WdT,   // (H,D): WdT[h][k] = Wd[k][h]
    const float* __restrict__ vd, float* __restrict__ SD)
{
  const int L = 256, Dd = 256;
  __shared__ __align__(16) float  hp[256];
  __shared__ __align__(16) float  vv[256];
  __shared__ __align__(16) bf16_t G[64][264];   // q-chunk x K (padded rows)
  __shared__ float  sdacc[64];

  int tid = threadIdx.x, lane = tid & 31, w = tid >> 5;
  int b = blockIdx.x >> 8;
  int p = blockIdx.x & 255;
  hp[tid] = Hp[((size_t)b * L + p) * Dd + tid];
  vv[tid] = vd[tid];
  const float* HqB = Hq + (size_t)b * L * Dd;
  int r = lane & 15, hh = lane >> 4;
  const bf16_t* pb0 = WdT + (size_t)(w * 32 + r) * Dd + hh * 16;
  const bf16_t* pb1 = pb0 + 16 * Dd;

  for (int qt = 0; qt < 4; ++qt) {
    __syncthreads();
    // G[q][k] = Hq[b, qt*64+q, k] * Hp[b,p,k]  (b128 loads, b64 bf16 stores)
    for (int idx = tid; idx < 64 * 64; idx += 256) {
      int q = idx >> 6, k4 = (idx & 63) << 2;
      v4f hq4 = *(const v4f*)&HqB[(size_t)(qt * 64 + q) * Dd + k4];
      v4f hp4 = *(const v4f*)&hp[k4];
      v4bf g;
      g.x = (bf16_t)(hq4.x * hp4.x);
      g.y = (bf16_t)(hq4.y * hp4.y);
      g.z = (bf16_t)(hq4.z * hp4.z);
      g.w = (bf16_t)(hq4.w * hp4.w);
      *(v4bf*)&G[q][k4] = g;
    }
    if (tid < 64) sdacc[tid] = 0.f;
    __syncthreads();

    v8f zero = {0.f,0.f,0.f,0.f,0.f,0.f,0.f,0.f};
    v8f acc[4][2];
#pragma unroll
    for (int mt = 0; mt < 4; ++mt) { acc[mt][0] = zero; acc[mt][1] = zero; }

    const bf16_t* pa0 = &G[r][hh * 16];   // + mt*16*264 + kt*32

#pragma unroll
    for (int kt = 0; kt < 8; ++kt) {
      FragU fa[4], fb[2];
#pragma unroll
      for (int mt = 0; mt < 4; ++mt) {
        const bf16_t* pa = pa0 + mt * 16 * 264 + kt * 32;
        fa[mt].h[0] = *(const v8bf*)pa;
        fa[mt].h[1] = *(const v8bf*)(pa + 8);
      }
      fb[0].h[0] = *(const v8bf*)(pb0 + kt * 32);
      fb[0].h[1] = *(const v8bf*)(pb0 + kt * 32 + 8);
      fb[1].h[0] = *(const v8bf*)(pb1 + kt * 32);
      fb[1].h[1] = *(const v8bf*)(pb1 + kt * 32 + 8);
#pragma unroll
      for (int mt = 0; mt < 4; ++mt) {
        acc[mt][0] = wmma_bf16(fa[mt].v, fb[0].v, acc[mt][0]);
        acc[mt][1] = wmma_bf16(fa[mt].v, fb[1].v, acc[mt][1]);
      }
    }

    // fused epilogue: sd[row] += sum over this wave's 32 h-columns
#pragma unroll
    for (int mt = 0; mt < 4; ++mt)
#pragma unroll
      for (int nt = 0; nt < 2; ++nt) {
        int col = w * 32 + nt * 16 + r;
        float vcol = vv[col];
#pragma unroll
        for (int v = 0; v < 8; ++v) {
          float t = tanh_hw(acc[mt][nt][v]) * vcol;
          t += __shfl_xor(t, 1, 16);
          t += __shfl_xor(t, 2, 16);
          t += __shfl_xor(t, 4, 16);
          t += __shfl_xor(t, 8, 16);
          if (r == 0)
            atomicAdd(&sdacc[mt * 16 + hh * 8 + v], t);
        }
      }
    __syncthreads();
    if (tid < 64)
      SD[((size_t)b * L + p) * L + qt * 64 + tid] = sdacc[tid];
  }
}

// ---------------------------------------------------------------------------
// Row softmax over 256 elements; writes bf16 attention weights.
// ---------------------------------------------------------------------------
__global__ __launch_bounds__(256) void softmax_rows_k(
    const float* __restrict__ S, bf16_t* __restrict__ A)
{
  __shared__ float red[256];
  int row = blockIdx.x, tid = threadIdx.x;
  float x = S[(size_t)row * 256 + tid];
  red[tid] = x;
  __syncthreads();
  for (int s = 128; s > 0; s >>= 1) {
    if (tid < s) red[tid] = fmaxf(red[tid], red[tid + s]);
    __syncthreads();
  }
  float m = red[0];
  __syncthreads();
  float e = __expf(x - m);
  red[tid] = e;
  __syncthreads();
  for (int s = 128; s > 0; s >>= 1) {
    if (tid < s) red[tid] += red[tid + s];
    __syncthreads();
  }
  float inv = 1.f / red[0];
  A[(size_t)row * 256 + tid] = (bf16_t)(e * inv);
}

// ---------------------------------------------------------------------------
extern "C" void kernel_launch(void* const* d_in, const int* in_sizes, int n_in,
                              void* d_out, int out_size, void* d_ws,
                              size_t ws_size, hipStream_t stream) {
  (void)in_sizes; (void)n_in; (void)out_size; (void)ws_size;
  const float* Hp  = (const float*)d_in[0];
  const float* Hq  = (const float*)d_in[1];
  const float* Wc1 = (const float*)d_in[2];
  const float* Wc2 = (const float*)d_in[3];
  const float* vc  = (const float*)d_in[4];
  const float* Wb  = (const float*)d_in[5];
  const float* Wd  = (const float*)d_in[6];
  const float* vd  = (const float*)d_in[7];
  const float* Wm  = (const float*)d_in[8];
  const float* vm  = (const float*)d_in[9];
  float* out = (float*)d_out;

  const size_t nHB = (size_t)4 * 256 * 256;  // 262144 per (B,L,D) tensor
  const size_t nW  = (size_t)256 * 256;      // 65536 per weight

  char* w = (char*)d_ws;
  auto carve = [&](size_t bytes) -> char* {
    char* q = w; w += (bytes + 255) & ~(size_t)255; return q;
  };
  bf16_t* HpB  = (bf16_t*)carve(nHB * 2);   // row-major bf16 (A operands)
  bf16_t* HqB  = (bf16_t*)carve(nHB * 2);
  bf16_t* HqT  = (bf16_t*)carve(nHB * 2);   // per-batch transposed (D x L)
  bf16_t* Wc1T = (bf16_t*)carve(nW * 2);    // (N x K) transposed weights
  bf16_t* Wc2T = (bf16_t*)carve(nW * 2);
  bf16_t* WmT  = (bf16_t*)carve(nW * 2);
  bf16_t* WbT  = (bf16_t*)carve(nW * 2);
  bf16_t* WdT  = (bf16_t*)carve(nW * 2);
  float*  s1   = (float*)carve(nHB * 4);
  float*  s2   = (float*)carve(nHB * 4);
  float*  m1   = (float*)carve(nHB * 4);
  float*  m2   = (float*)carve(nHB * 4);
  bf16_t* HqbB = (bf16_t*)carve(nHB * 2);
  float*  sc   = (float*)carve(nHB * 4);
  float*  sb   = (float*)carve(nHB * 4);
  float*  sd   = (float*)carve(nHB * 4);
  float*  sm   = (float*)carve(nHB * 4);
  bf16_t* aB   = (bf16_t*)carve(nHB * 2);

  // 1) convert / transpose operands to bf16
  cvt_bf16_k<<<(int)(nHB / 1024), 256, 0, stream>>>(Hp, HpB, (int)(nHB / 4));
  cvt_bf16_k<<<(int)(nHB / 1024), 256, 0, stream>>>(Hq, HqB, (int)(nHB / 4));
  cvt_t_bf16_k<<<64, 256, 0, stream>>>(Wc1, Wc1T, 256, 256, 0L, 0L);
  cvt_t_bf16_k<<<64, 256, 0, stream>>>(Wc2, Wc2T, 256, 256, 0L, 0L);
  cvt_t_bf16_k<<<64, 256, 0, stream>>>(Wm,  WmT,  256, 256, 0L, 0L);
  cvt_t_bf16_k<<<64, 256, 0, stream>>>(Wb,  WbT,  256, 256, 0L, 0L);
  cvt_t_bf16_k<<<64, 256, 0, stream>>>(Wd,  WdT,  256, 256, 0L, 0L);
  cvt_t_bf16_k<<<256, 256, 0, stream>>>(Hq, HqT, 256, 256, 65536L, 65536L);

  // 2) projections: (B*L=1024, 256) @ (256, 256) -> 64x16 tiles = 1024 waves
  gemm_bt_k<float><<<128, 256, 0, stream>>>(HqB, Wc1T, s1, 64, 16, 1,
                                            256, 256, 256, 0L, 0L, 0L);
  gemm_bt_k<float><<<128, 256, 0, stream>>>(HpB, Wc2T, s2, 64, 16, 1,
                                            256, 256, 256, 0L, 0L, 0L);
  gemm_bt_k<float><<<128, 256, 0, stream>>>(HqB, WmT, m1, 64, 16, 1,
                                            256, 256, 256, 0L, 0L, 0L);
  gemm_bt_k<float><<<128, 256, 0, stream>>>(HpB, WmT, m2, 64, 16, 1,
                                            256, 256, 256, 0L, 0L, 0L);
  gemm_bt_k<bf16_t><<<128, 256, 0, stream>>>(HqB, WbT, HqbB, 64, 16, 1,
                                             256, 256, 256, 0L, 0L, 0L);

  // 3) scores
  addsub_scores_k<<<1024, 256, 0, stream>>>(s1, s2, vc, 1.0f, sc);
  addsub_scores_k<<<1024, 256, 0, stream>>>(m1, m2, vm, -1.0f, sm);
  // sb[b] = Hp[b] @ (Hq Wb)[b]^T : Bt = Hqb rows are exactly B columns
  gemm_bt_k<float><<<128, 256, 0, stream>>>(HpB, HqbB, sb, 16, 16, 4,
                                            256, 256, 256, 65536L, 65536L, 65536L);
  branch_d_scores_k<<<1024, 256, 0, stream>>>(Hq, Hp, WdT, vd, sd);

  // 4) softmax + attend per branch (outputs: qc, qb, qd, qm)
  const float* scores[4] = {sc, sb, sd, sm};
  for (int br = 0; br < 4; ++br) {
    softmax_rows_k<<<1024, 256, 0, stream>>>(scores[br], aB);
    gemm_bt_k<float><<<128, 256, 0, stream>>>(aB, HqT, out + (size_t)br * nHB,
                                              16, 16, 4,
                                              256, 256, 256, 65536L, 65536L, 65536L);
  }
}
